// GraphMaskedLM_86208583565974
// MI455X (gfx1250) — matrix operand order repaired and checked by reading, not compiled, not run on hardware
//
#include <hip/hip_runtime.h>
#include <hip/hip_bf16.h>
#include <cstdint>
#include <climits>

// Problem constants (from reference)
#define IN_DIM 256
#define HIDC   64
#define HH1    4
#define HH2    2
#define BHD    768
#define FHD    512
#define NCLS   100
#define EPSV   1e-5f

typedef __attribute__((ext_vector_type(16))) _Float16 v16h;
typedef __attribute__((ext_vector_type(8)))  _Float16 v8h;
typedef __attribute__((ext_vector_type(8)))  float    v8f;

// ---------------- utility ----------------
__device__ __forceinline__ int fkey(float x) {
  int b = __float_as_int(x);
  return b >= 0 ? b : (b ^ 0x7fffffff);   // order-preserving int key for floats
}
__device__ __forceinline__ float funkey(int k) {
  return __int_as_float(k >= 0 ? k : (k ^ 0x7fffffff));
}

__global__ void fill_f32(float* __restrict__ p, float v, long n) {
  long i = (long)blockIdx.x * blockDim.x + threadIdx.x;
  if (i < n) p[i] = v;
}
__global__ void fill_i32(int* __restrict__ p, int v, long n) {
  long i = (long)blockIdx.x * blockDim.x + threadIdx.x;
  if (i < n) p[i] = v;
}

// transpose + convert weight: Wt[n][k] (f16) = W[k][n] (f32), K x Nc row-major in.
__global__ void conv_tr_f16(const float* __restrict__ W, _Float16* __restrict__ Wt,
                            int K, int Nc)
{
  long i = (long)blockIdx.x * blockDim.x + threadIdx.x;
  if (i >= (long)K * Nc) return;
  int n = (int)(i / K), k = (int)(i % K);
  Wt[i] = (_Float16)W[(long)k * Nc + n];
}

__device__ __forceinline__ v16h load_bfrag(const _Float16* p) {
  v8h lo = *(const v8h*)(p);
  v8h hh = *(const v8h*)(p + 8);
  v16h bf;
#pragma unroll
  for (int i = 0; i < 8; ++i) { bf[i] = lo[i]; bf[8 + i] = hh[i]; }
  return bf;
}

__device__ __forceinline__ v16h load_afrag(const float* ap) {
  v16h af;
#pragma unroll
  for (int i = 0; i < 8; ++i) af[i] = (_Float16)ap[i];
#pragma unroll
  for (int i = 0; i < 8; ++i) af[8 + i] = (_Float16)ap[8 + i];
  return af;
}

// ---------------- WMMA GEMM: C = act(A@Bt^T + bias + rowbias[rowsel]) ----------------
// A: [M,K] f32 (lda). Bt: pre-transposed+converted weight, f16 [Nc][K] row-major.
// C: [M,Nc] f32 (ldc). One wave -> 16x64 output strip: 4 WMMA tiles sharing one
// A fragment; K stepped by 32 with v_wmma_f32_16x16x32_f16, f32 accumulation.
// The K-loop is software-pipelined: next iteration's fragments are loaded before
// the current WMMAs consume theirs, forcing distinct registers + overlapped waits.
template<int ACT>  // 0 = none, 1 = relu
__global__ __launch_bounds__(256) void gemm_wmma(
    const float* __restrict__ A, const _Float16* __restrict__ Bt,
    const float* __restrict__ bias,            // [Nc] or null
    const float* __restrict__ rowbias,         // [2,Nc] or null
    const unsigned char* __restrict__ rowsel,  // [M]  or null
    float* __restrict__ C,
    int M, int K, int Nc, int lda, int ldc)
{
  const int lane = threadIdx.x & 31;
  const int wave = threadIdx.x >> 5;
  const int r  = lane & 15;
  const int hi = lane >> 4;
  const int tiles_n = (Nc + 63) >> 6;          // 64-column groups
  const int tiles_m = (M + 15) >> 4;
  long tile = (long)blockIdx.x * 8 + wave;
  if (tile >= (long)tiles_m * tiles_n) return; // uniform per wave
  const int tile_m = (int)(tile / tiles_n);
  const int tile_n = (int)(tile % tiles_n);

  int arow = tile_m * 16 + r;
  if (arow >= M) arow = M - 1;                 // clamp (stores are guarded)
  const float* arp = A + (long)arow * lda + 16 * hi;

  // per-tile loop-invariant B base pointers (col-major f16 rows of length K)
  const _Float16* bb0;
  const _Float16* bb1;
  const _Float16* bb2;
  const _Float16* bb3;
  {
    int c0 = tile_n * 64 + r;       if (c0 >= Nc) c0 = Nc - 1;
    int c1 = tile_n * 64 + 16 + r;  if (c1 >= Nc) c1 = Nc - 1;
    int c2 = tile_n * 64 + 32 + r;  if (c2 >= Nc) c2 = Nc - 1;
    int c3 = tile_n * 64 + 48 + r;  if (c3 >= Nc) c3 = Nc - 1;
    bb0 = Bt + (size_t)c0 * K + 16 * hi;
    bb1 = Bt + (size_t)c1 * K + 16 * hi;
    bb2 = Bt + (size_t)c2 * K + 16 * hi;
    bb3 = Bt + (size_t)c3 * K + 16 * hi;
  }

  v8f acc0 = {}, acc1 = {}, acc2 = {}, acc3 = {};

  // ---- prologue: fragments for k=0
  v16h af  = load_afrag(arp);
  v16h bf0 = load_bfrag(bb0);
  v16h bf1 = load_bfrag(bb1);
  v16h bf2 = load_bfrag(bb2);
  v16h bf3 = load_bfrag(bb3);

  // ---- steady state: load k0's fragments, compute with (k0-32)'s
  for (int k0 = 32; k0 < K; k0 += 32) {
    v16h afn = load_afrag(arp + k0);
    v16h b0n = load_bfrag(bb0 + k0);
    v16h b1n = load_bfrag(bb1 + k0);
    v16h b2n = load_bfrag(bb2 + k0);
    v16h b3n = load_bfrag(bb3 + k0);

    acc0 = __builtin_amdgcn_wmma_f32_16x16x32_f16(false, af, false, bf0, (short)0, acc0, false, false);
    acc1 = __builtin_amdgcn_wmma_f32_16x16x32_f16(false, af, false, bf1, (short)0, acc1, false, false);
    acc2 = __builtin_amdgcn_wmma_f32_16x16x32_f16(false, af, false, bf2, (short)0, acc2, false, false);
    acc3 = __builtin_amdgcn_wmma_f32_16x16x32_f16(false, af, false, bf3, (short)0, acc3, false, false);

    af = afn; bf0 = b0n; bf1 = b1n; bf2 = b2n; bf3 = b3n;
  }

  // ---- epilogue: last K block
  acc0 = __builtin_amdgcn_wmma_f32_16x16x32_f16(false, af, false, bf0, (short)0, acc0, false, false);
  acc1 = __builtin_amdgcn_wmma_f32_16x16x32_f16(false, af, false, bf1, (short)0, acc1, false, false);
  acc2 = __builtin_amdgcn_wmma_f32_16x16x32_f16(false, af, false, bf2, (short)0, acc2, false, false);
  acc3 = __builtin_amdgcn_wmma_f32_16x16x32_f16(false, af, false, bf3, (short)0, acc3, false, false);

  // ---- epilogue: D layout: VGPR v -> row m = tile_m*16 + v + 8*hi, col = group + r
  v8f accs[4] = {acc0, acc1, acc2, acc3};
#pragma unroll
  for (int t = 0; t < 4; ++t) {
    int col = tile_n * 64 + t * 16 + r;
    if (col >= Nc) continue;
    float bv = bias ? bias[col] : 0.0f;
#pragma unroll
    for (int v = 0; v < 8; ++v) {
      int m = tile_m * 16 + v + 8 * hi;
      if (m >= M) continue;
      float val = accs[t][v] + bv;
      if (rowbias) val += rowbias[(rowsel[m] ? 1 : 0) * Nc + col];
      if (ACT == 1) val = val > 0.0f ? val : 0.0f;
      C[(long)m * ldc + col] = val;
    }
  }
}

// ---------------- GAT pieces ----------------
__global__ void gat_scores(const float* __restrict__ h,
                           const float* __restrict__ a_src,
                           const float* __restrict__ a_dst,
                           float* __restrict__ s, float* __restrict__ d,
                           int Nn, int H, int Cc, int ld)
{
  int idx = blockIdx.x * blockDim.x + threadIdx.x;
  if (idx >= Nn * H) return;
  int n = idx / H, hh = idx % H;
  const float* hp = h + (long)n * ld + hh * Cc;
  const float* as = a_src + hh * Cc;
  const float* ad = a_dst + hh * Cc;
  float ss = 0.f, dd = 0.f;
  for (int c = 0; c < Cc; ++c) { float v = hp[c]; ss += v * as[c]; dd += v * ad[c]; }
  s[idx] = ss; d[idx] = dd;
}

__global__ void edge_max(const int* __restrict__ src, const int* __restrict__ dst,
                         int E, int Nn, const float* __restrict__ s,
                         const float* __restrict__ d, int H, int* __restrict__ mkey)
{
  int e = blockIdx.x * blockDim.x + threadIdx.x;
  int tot = E + Nn;
  if (e >= tot) return;
  int se = e < E ? src[e] : (e - E);
  int de = e < E ? dst[e] : (e - E);
  for (int h = 0; h < H; ++h) {
    float x = s[se * H + h] + d[de * H + h];
    x = x > 0.f ? x : 0.2f * x;            // leaky_relu 0.2
    atomicMax(&mkey[de * H + h], fkey(x));
  }
}

__global__ void edge_den(const int* __restrict__ src, const int* __restrict__ dst,
                         int E, int Nn, const float* __restrict__ s,
                         const float* __restrict__ d, int H,
                         const int* __restrict__ mkey, float* __restrict__ den)
{
  int e = blockIdx.x * blockDim.x + threadIdx.x;
  int tot = E + Nn;
  if (e >= tot) return;
  int se = e < E ? src[e] : (e - E);
  int de = e < E ? dst[e] : (e - E);
  for (int h = 0; h < H; ++h) {
    float x = s[se * H + h] + d[de * H + h];
    x = x > 0.f ? x : 0.2f * x;
    float m = funkey(mkey[de * H + h]);
    atomicAdd(&den[de * H + h], __expf(x - m));
  }
}

__global__ void edge_agg(const int* __restrict__ src, const int* __restrict__ dst,
                         int E, int Nn, const float* __restrict__ s,
                         const float* __restrict__ d, int H, int Cc,
                         const int* __restrict__ mkey, const float* __restrict__ den,
                         const float* __restrict__ hlin, int ld,
                         float* __restrict__ outp, int ldo)
{
  int idx = blockIdx.x * blockDim.x + threadIdx.x;
  int tot = (E + Nn) * H;
  if (idx >= tot) return;
  int e = idx / H, h = idx % H;
  int se = e < E ? src[e] : (e - E);
  int de = e < E ? dst[e] : (e - E);
  float x = s[se * H + h] + d[de * H + h];
  x = x > 0.f ? x : 0.2f * x;
  float m = funkey(mkey[de * H + h]);
  float a = __expf(x - m) / den[de * H + h];
  const float* hs = hlin + (long)se * ld + h * Cc;
  float* o = outp + (long)de * ldo + h * Cc;
  for (int c = 0; c < Cc; ++c) atomicAdd(&o[c], hs[c] * a);
}

// ---------------- BatchNorm (column stats over N) ----------------
#define CS_ROWS 512
__global__ __launch_bounds__(256) void col_stats(const float* __restrict__ X,
                                                 const float* __restrict__ bias,
                                                 int Nn, int cols, int ld,
                                                 float* __restrict__ sums) // [2*cols]
{
  int c = threadIdx.x;
  if (c >= cols) return;
  int r0 = blockIdx.x * CS_ROWS;
  int r1 = r0 + CS_ROWS; if (r1 > Nn) r1 = Nn;
  float bv = bias[c], s = 0.f, q = 0.f;
  for (int r = r0; r < r1; ++r) { float v = X[(long)r * ld + c] + bv; s += v; q += v * v; }
  atomicAdd(&sums[c], s);
  atomicAdd(&sums[cols + c], q);
}

__global__ void bn_elu(float* __restrict__ X, const float* __restrict__ bias,
                       const float* __restrict__ sums, const float* __restrict__ g,
                       const float* __restrict__ b, int Nn, int cols, int ld)
{
  long i = (long)blockIdx.x * blockDim.x + threadIdx.x;
  if (i >= (long)Nn * cols) return;
  int c = (int)(i % cols);
  long r = i / cols;
  float mu  = sums[c] / (float)Nn;
  float var = sums[cols + c] / (float)Nn - mu * mu;
  float v = X[r * ld + c] + bias[c];
  v = (v - mu) * rsqrtf(var + EPSV) * g[c] + b[c];
  v = v > 0.f ? v : (__expf(v) - 1.0f);   // elu(alpha=1)
  X[r * ld + c] = v;
}

// ---------------- LayerNorm over last dim (one block per row) ----------------
template<int RELU>
__global__ __launch_bounds__(256) void ln_rows(float* __restrict__ X,
                                               const float* __restrict__ g,
                                               const float* __restrict__ b,
                                               int Nn, int cols, int ld)
{
  __shared__ float ss[256], qq[256];
  int row = blockIdx.x;
  if (row >= Nn) return;
  float* xp = X + (long)row * ld;
  float s = 0.f, q = 0.f;
  for (int c = threadIdx.x; c < cols; c += 256) { float v = xp[c]; s += v; q += v * v; }
  ss[threadIdx.x] = s; qq[threadIdx.x] = q;
  __syncthreads();
  for (int o = 128; o > 0; o >>= 1) {
    if ((int)threadIdx.x < o) { ss[threadIdx.x] += ss[threadIdx.x + o]; qq[threadIdx.x] += qq[threadIdx.x + o]; }
    __syncthreads();
  }
  float mu  = ss[0] / (float)cols;
  float var = qq[0] / (float)cols - mu * mu;
  float ri  = rsqrtf(var + EPSV);
  for (int c = threadIdx.x; c < cols; c += 256) {
    float v = (xp[c] - mu) * ri * g[c] + b[c];
    if (RELU) v = v > 0.f ? v : 0.f;
    xp[c] = v;
  }
}

// fold bert_table @ Wf1[768:,:] + bf1 into a 2x512 row-bias table
__global__ void bert_fold(const float* __restrict__ bert_table,
                          const float* __restrict__ Wf1,
                          const float* __restrict__ bf1,
                          float* __restrict__ t01)
{
  int idx = blockIdx.x * blockDim.x + threadIdx.x;
  if (idx >= 2 * FHD) return;
  int v = idx / FHD, j = idx % FHD;
  const float* bt = bert_table + v * BHD;
  float acc = bf1[j];
  for (int k = 0; k < BHD; ++k) acc += bt[k] * Wf1[(long)(BHD + k) * FHD + j];
  t01[idx] = acc;
}

// ---------------- host-side helpers ----------------
static inline void launch_gemm(hipStream_t st, const float* A, const _Float16* Bt,
                               const float* bias, const float* rowbias,
                               const unsigned char* rowsel, float* C,
                               int M, int K, int Nc, int lda, int ldc, int act)
{
  int tiles_m = (M + 15) / 16, tiles_n = (Nc + 63) / 64;
  long tiles = (long)tiles_m * tiles_n;
  int blocks = (int)((tiles + 7) / 8);
  if (act == 1)
    gemm_wmma<1><<<blocks, 256, 0, st>>>(A, Bt, bias, rowbias, rowsel, C, M, K, Nc, lda, ldc);
  else
    gemm_wmma<0><<<blocks, 256, 0, st>>>(A, Bt, bias, rowbias, rowsel, C, M, K, Nc, lda, ldc);
}
static inline int cdiv(long a, long b) { return (int)((a + b - 1) / b); }

extern "C" void kernel_launch(void* const* d_in, const int* in_sizes, int n_in,
                              void* d_out, int out_size, void* d_ws, size_t ws_size,
                              hipStream_t stream)
{
  if (n_in < 30) return;
  const float* x     = (const float*)d_in[0];
  const int*   ei    = (const int*)d_in[1];
  const unsigned char* mask = (const unsigned char*)d_in[2];
  const float* W1    = (const float*)d_in[3];
  const float* as1   = (const float*)d_in[4];
  const float* ad1   = (const float*)d_in[5];
  const float* b1    = (const float*)d_in[6];
  const float* gbn1  = (const float*)d_in[7];
  const float* bbn1  = (const float*)d_in[8];
  const float* W2    = (const float*)d_in[9];
  const float* as2   = (const float*)d_in[10];
  const float* ad2   = (const float*)d_in[11];
  const float* b2    = (const float*)d_in[12];
  const float* gbn2  = (const float*)d_in[13];
  const float* bbn2  = (const float*)d_in[14];
  const float* Wp    = (const float*)d_in[15];
  const float* bp    = (const float*)d_in[16];
  const float* btab  = (const float*)d_in[17];
  const float* Wf1   = (const float*)d_in[18];
  const float* bf1   = (const float*)d_in[19];
  const float* gln1  = (const float*)d_in[20];
  const float* bln1  = (const float*)d_in[21];
  const float* Wf2   = (const float*)d_in[22];
  const float* bf2   = (const float*)d_in[23];
  const float* gln2  = (const float*)d_in[24];
  const float* bln2  = (const float*)d_in[25];
  const float* Wc1   = (const float*)d_in[26];
  const float* bc1   = (const float*)d_in[27];
  const float* Wc2   = (const float*)d_in[28];
  const float* bc2   = (const float*)d_in[29];

  const int N = in_sizes[0] / IN_DIM;
  const int E = in_sizes[1] / 2;
  const int* src = ei;
  const int* dst = ei + E;

  // workspace (floats): A:N*768 | B:N*512 | C:N*128 | s,d:N*4 | mkey:N*4(int) | den:N*4 | sums:1536 | t01:1024 | f16 weights
  float* Abuf = (float*)d_ws;
  float* Bbuf = Abuf + (size_t)N * 768;
  float* Cbuf = Bbuf + (size_t)N * 512;
  float* sbuf = Cbuf + (size_t)N * 128;
  float* dbuf = sbuf + (size_t)N * 4;
  int*   mkey = (int*)(dbuf + (size_t)N * 4);
  float* den  = (float*)(mkey + (size_t)N * 4);
  float* sums = den + (size_t)N * 4;
  float* t01  = sums + 1536;
  // f16 transposed weights (16B-aligned: all preceding sizes are multiples of 4 floats)
  _Float16* W1t  = (_Float16*)(t01 + 1024);
  _Float16* W2t  = W1t  + (size_t)256 * 256;   // [256][256]
  _Float16* Wpt  = W2t  + (size_t)128 * 256;   // [128][256]
  _Float16* Wf1t = Wpt  + (size_t)768 * 128;   // [768][128]
  _Float16* Wf2t = Wf1t + (size_t)512 * 768;   // [512][768]
  _Float16* Wc1t = Wf2t + (size_t)768 * 512;   // [768][512]
  _Float16* Wc2t = Wc1t + (size_t)64  * 768;   // [64][768]
                                               // Wc2t: [100][64]

  const int TB = 256;

  // ===== one-time: transpose+convert all weights to f16 [Nc][K] =====
  conv_tr_f16<<<cdiv(256L * 256, TB), TB, 0, stream>>>(W1,  W1t,  256, 256);
  conv_tr_f16<<<cdiv(256L * 128, TB), TB, 0, stream>>>(W2,  W2t,  256, 128);
  conv_tr_f16<<<cdiv(128L * 768, TB), TB, 0, stream>>>(Wp,  Wpt,  128, 768);
  conv_tr_f16<<<cdiv(768L * 512, TB), TB, 0, stream>>>(Wf1, Wf1t, 768, 512);  // top 768 rows only
  conv_tr_f16<<<cdiv(512L * 768, TB), TB, 0, stream>>>(Wf2, Wf2t, 512, 768);
  conv_tr_f16<<<cdiv(768L * 64,  TB), TB, 0, stream>>>(Wc1, Wc1t, 768, 64);
  conv_tr_f16<<<cdiv(64L * 100,  TB), TB, 0, stream>>>(Wc2, Wc2t, 64,  100);

  // ===== GAT layer 1 (H=4, C=64, feat 256) =====
  launch_gemm(stream, x, W1t, nullptr, nullptr, nullptr, Abuf, N, IN_DIM, 256, IN_DIM, 256, 0);
  gat_scores<<<cdiv((long)N * HH1, TB), TB, 0, stream>>>(Abuf, as1, ad1, sbuf, dbuf, N, HH1, HIDC, 256);
  fill_i32<<<cdiv((long)N * HH1, TB), TB, 0, stream>>>(mkey, INT_MIN, (long)N * HH1);
  fill_f32<<<cdiv((long)N * HH1, TB), TB, 0, stream>>>(den, 0.f, (long)N * HH1);
  fill_f32<<<cdiv((long)N * 256, TB), TB, 0, stream>>>(Bbuf, 0.f, (long)N * 256);
  fill_f32<<<cdiv(1536L, TB), TB, 0, stream>>>(sums, 0.f, 1536L);
  edge_max<<<cdiv((long)(E + N), TB), TB, 0, stream>>>(src, dst, E, N, sbuf, dbuf, HH1, mkey);
  edge_den<<<cdiv((long)(E + N), TB), TB, 0, stream>>>(src, dst, E, N, sbuf, dbuf, HH1, mkey, den);
  edge_agg<<<cdiv((long)(E + N) * HH1, TB), TB, 0, stream>>>(src, dst, E, N, sbuf, dbuf, HH1, HIDC,
                                                             mkey, den, Abuf, 256, Bbuf, 256);
  col_stats<<<cdiv((long)N, CS_ROWS), TB, 0, stream>>>(Bbuf, b1, N, 256, 256, sums);
  bn_elu<<<cdiv((long)N * 256, TB), TB, 0, stream>>>(Bbuf, b1, sums, gbn1, bbn1, N, 256, 256);

  // ===== GAT layer 2 (H=2, C=64, feat 128) =====
  launch_gemm(stream, Bbuf, W2t, nullptr, nullptr, nullptr, Abuf, N, 256, 128, 256, 128, 0);
  gat_scores<<<cdiv((long)N * HH2, TB), TB, 0, stream>>>(Abuf, as2, ad2, sbuf, dbuf, N, HH2, HIDC, 128);
  fill_i32<<<cdiv((long)N * HH2, TB), TB, 0, stream>>>(mkey, INT_MIN, (long)N * HH2);
  fill_f32<<<cdiv((long)N * HH2, TB), TB, 0, stream>>>(den, 0.f, (long)N * HH2);
  fill_f32<<<cdiv((long)N * 128, TB), TB, 0, stream>>>(Cbuf, 0.f, (long)N * 128);
  fill_f32<<<cdiv(1536L, TB), TB, 0, stream>>>(sums, 0.f, 1536L);
  edge_max<<<cdiv((long)(E + N), TB), TB, 0, stream>>>(src, dst, E, N, sbuf, dbuf, HH2, mkey);
  edge_den<<<cdiv((long)(E + N), TB), TB, 0, stream>>>(src, dst, E, N, sbuf, dbuf, HH2, mkey, den);
  edge_agg<<<cdiv((long)(E + N) * HH2, TB), TB, 0, stream>>>(src, dst, E, N, sbuf, dbuf, HH2, HIDC,
                                                             mkey, den, Abuf, 128, Cbuf, 128);
  col_stats<<<cdiv((long)N, CS_ROWS), TB, 0, stream>>>(Cbuf, b2, N, 128, 128, sums);
  bn_elu<<<cdiv((long)N * 128, TB), TB, 0, stream>>>(Cbuf, b2, sums, gbn2, bbn2, N, 128, 128);

  // ===== projection to 768 =====
  launch_gemm(stream, Cbuf, Wpt, bp, nullptr, nullptr, Abuf, N, 128, BHD, 128, BHD, 0);

  // ===== fold bert branch of Wf1, then f1 = relu(LN(...)) =====
  bert_fold<<<cdiv(2L * FHD, TB), TB, 0, stream>>>(btab, Wf1, bf1, t01);
  launch_gemm(stream, Abuf, Wf1t, nullptr, t01, mask, Bbuf, N, BHD, FHD, BHD, FHD, 0);
  ln_rows<1><<<N, 256, 0, stream>>>(Bbuf, gln1, bln1, N, FHD, FHD);

  // ===== f2 = LN(f1 @ Wf2 + bf2) =====
  launch_gemm(stream, Bbuf, Wf2t, bf2, nullptr, nullptr, Abuf, N, FHD, BHD, FHD, BHD, 0);
  ln_rows<0><<<N, 256, 0, stream>>>(Abuf, gln2, bln2, N, BHD, BHD);

  // ===== classifier: relu(f2 @ Wc1 + bc1) @ Wc2 + bc2 =====
  launch_gemm(stream, Abuf, Wc1t, bc1, nullptr, nullptr, Cbuf, N, BHD, HIDC, BHD, HIDC, 1);
  launch_gemm(stream, Cbuf, Wc2t, bc2, nullptr, nullptr, (float*)d_out, N, HIDC, NCLS, HIDC, NCLS, 0);
}
